// MSSTF_38130719654501
// MI455X (gfx1250) — compile-verified, matrix operand order checked by reading, not run on hardware
//
#include <hip/hip_runtime.h>
#include <hip/hip_bf16.h>
#include <math.h>

// ---------------- problem constants (from reference) ----------------
#define DMODEL 16
#define HH     24
#define WWID   24
#define NPIX   576          // 24*24
#define TFULL  32
#define BB     2
#define WINL   16
#define NWINP  17           // TFULL - WINL + 1

typedef __attribute__((ext_vector_type(16))) _Float16 v16h;
typedef __attribute__((ext_vector_type(8)))  float    v8f;

__device__ __forceinline__ int imin(int a, int b) { return a < b ? a : b; }
__device__ __forceinline__ int imax(int a, int b) { return a > b ? a : b; }

// ---------------- WMMA helpers (v_wmma_f32_16x16x32_f16, K=16 used) -------
// A: 16xK row-major (m = lane&15, K striped per half-wave per ISA 7.12.2)
__device__ __forceinline__ v16h pack_a(const float* A, int lda, int mR, int kC) {
    const int lane = threadIdx.x & 31;
    const int m = lane & 15, kb = (lane >> 4) * 8;
    v16h a;
#pragma unroll
    for (int i = 0; i < 8; ++i) {
        const int k = kb + i;
        a[i]     = (_Float16)((m < mR && k < kC) ? A[m * lda + k] : 0.f);
        a[i + 8] = (_Float16)0.f;           // K = 16..31 zero padded
    }
    return a;
}
// B: KxN row-major in memory (B[k*ldb + n])
__device__ __forceinline__ v16h pack_b(const float* B, int ldb, int kC, int nC) {
    const int lane = threadIdx.x & 31;
    const int n = lane & 15, kb = (lane >> 4) * 8;
    v16h b;
#pragma unroll
    for (int i = 0; i < 8; ++i) {
        const int k = kb + i;
        b[i]     = (_Float16)((n < nC && k < kC) ? B[k * ldb + n] : 0.f);
        b[i + 8] = (_Float16)0.f;
    }
    return b;
}
// B given transposed: Bmat[k][n] = Bt[n*ldb + k]  (weights stored (out,in))
__device__ __forceinline__ v16h pack_bT(const float* Bt, int ldb, int kC) {
    const int lane = threadIdx.x & 31;
    const int n = lane & 15, kb = (lane >> 4) * 8;
    v16h b;
#pragma unroll
    for (int i = 0; i < 8; ++i) {
        const int k = kb + i;
        b[i]     = (_Float16)((k < kC) ? Bt[n * ldb + k] : 0.f);
        b[i + 8] = (_Float16)0.f;
    }
    return b;
}
__device__ __forceinline__ v8f wmma_mac(v16h a, v16h b, v8f c) {
    return __builtin_amdgcn_wmma_f32_16x16x32_f16(false, a, false, b, (short)0, c,
                                                  false, false);
}
__device__ __forceinline__ v8f wmma_gemm(const float* A, int lda, int mR, int kC,
                                         const float* B, int ldb, int nC, v8f c) {
    return wmma_mac(pack_a(A, lda, mR, kC), pack_b(B, ldb, kC, nC), c);
}
__device__ __forceinline__ v8f wmma_gemmT(const float* A, int lda, int mR, int kC,
                                          const float* Bt, int ldb, v8f c) {
    return wmma_mac(pack_a(A, lda, mR, kC), pack_bT(Bt, ldb, kC), c);
}
// stores: C/D layout m = r + 8*(lane>=16), n = lane&15
__device__ __forceinline__ void wmma_store_bm(float* D, int ldd, int mR, v8f c,
                                              const float* bias_m) {
    const int lane = threadIdx.x & 31, n = lane & 15, half = lane >> 4;
#pragma unroll
    for (int r = 0; r < 8; ++r) {
        const int m = r + 8 * half;
        if (m < mR) D[m * ldd + n] = c[r] + (bias_m ? bias_m[m] : 0.f);
    }
}
__device__ __forceinline__ void wmma_store_bn(float* D, int ldd, int nC, v8f c,
                                              const float* bias_n) {
    const int lane = threadIdx.x & 31, n = lane & 15, half = lane >> 4;
    if (n >= nC) return;
#pragma unroll
    for (int r = 0; r < 8; ++r)
        D[(r + 8 * half) * ldd + n] = c[r] + (bias_n ? bias_n[n] : 0.f);
}
__device__ __forceinline__ void wmma_store_scale(float* D, int ldd, v8f c, float s) {
    const int lane = threadIdx.x & 31, n = lane & 15, half = lane >> 4;
#pragma unroll
    for (int r = 0; r < 8; ++r) D[(r + 8 * half) * ldd + n] = c[r] * s;
}

__device__ __forceinline__ float gelu_exact(float v) {
    return 0.5f * v * (1.f + erff(v * 0.70710678118654752f));
}

// ============================================================================
// attnconv: one workgroup (256 thr = 8 waves) per frame (16 ch x 24x24)
// MODE 0: frame f -> (b = f/32, t = f%32)          [xg path, 64 frames]
// MODE 1: frame f -> piece (i,b,w), t = i+w        [piece paths, 544 frames]
// ============================================================================
struct AcP {
    const float* emb_w;                              // 16x16
    const float* qkv_w[3]; const float* qkv_b[3];    // 12x16, 12
    const float* mix_w[3]; const float* mix_b[3];    // 12*k*k, 12
    const float* a1w[3];   const float* a1b[3];      // 4x4, 4
    const float* a2w[3];   const float* a2b[3];      // 4x4, 4
    const float* gq_w;  const float* gq_b;           // 4x16, 4
    const float* gkv_w; const float* gkv_b;          // 8x16, 8
    const float* proj_w; const float* proj_b;        // 16x16, 16
};

template <int MODE>
__global__ __launch_bounds__(256) void attnconv_kernel(
    const float* __restrict__ x, float* __restrict__ out, AcP P, int k0, int k1,
    int k2) {
    extern __shared__ float smem[];
    float* sx = smem;                 // 16*576: input, later res concat
    float* sy = sx + 16 * NPIX;       // 16*576: emb output
    float* sq = sy + 16 * NPIX;       // 12*576: qkv pre-mix / gq
    float* sm = sq + 12 * NPIX;       // 12*576: mix out / pool / kv

    const int f = blockIdx.x;
    int b, t;
    if (MODE == 0) {
        b = f / TFULL; t = f % TFULL;
    } else {
        const int i = f / (BB * WINL), r = f % (BB * WINL);
        b = r / WINL; t = i + (r % WINL);
    }
    const int tid = threadIdx.x, wave = tid >> 5;

    // ---- stage input frame into LDS ----
    for (int idx = tid; idx < 16 * NPIX; idx += 256) {
        const int c = idx / NPIX, p = idx % NPIX;
        const float* src = x + (((size_t)(b * 16 + c)) * TFULL + t) * NPIX + p;
        __builtin_prefetch(src + 256, 0, 1);
        sx[idx] = *src;
    }
    __syncthreads();

    // ---- emb: sy = emb_w(16x16) @ sx(16x576), WMMA per 16-pixel tile ----
    for (int tile = wave; tile < 36; tile += 8) {
        v8f c = {};
        c = wmma_gemm(P.emb_w, 16, 16, 16, sx + tile * 16, NPIX, 16, c);
        wmma_store_bm(sy + tile * 16, NPIX, 16, c, nullptr);
    }
    __syncthreads();

    const float scal = 0.70710678118654752f;  // DH^-0.5

    // ---- three local gated-conv branches ----
    for (int j = 0; j < 3; ++j) {
        const int kk = (j == 0) ? k0 : ((j == 1) ? k1 : k2);
        const int pad = kk / 2;
        // qkv (12x16 weights, padded to 16 rows) via WMMA
        for (int tile = wave; tile < 36; tile += 8) {
            v8f c = {};
            c = wmma_gemm(P.qkv_w[j], 16, 12, 16, sy + tile * 16, NPIX, 16, c);
            wmma_store_bm(sq + tile * 16, NPIX, 12, c, P.qkv_b[j]);
        }
        __syncthreads();
        // depthwise kxk conv with edge padding (VALU)
        for (int idx = tid; idx < 12 * NPIX; idx += 256) {
            const int c = idx / NPIX, p = idx % NPIX, h = p / WWID, w = p % WWID;
            float s = 0.f;
            for (int i = 0; i < kk; ++i) {
                const int hh = imin(imax(h + i - pad, 0), HH - 1);
                for (int jj = 0; jj < kk; ++jj) {
                    const int ww = imin(imax(w + jj - pad, 0), WWID - 1);
                    s += P.mix_w[j][(c * kk + i) * kk + jj] *
                         sq[c * NPIX + hh * WWID + ww];
                }
            }
            sm[idx] = s + P.mix_b[j][c];
        }
        __syncthreads();
        // gating: a=act1(q*k); a=a*sig(a); a=act2(a); res=tanh(a*scal)*v (4ch)
        for (int p = tid; p < NPIX; p += 256) {
            float tq[4], a1[4], a2[4];
#pragma unroll
            for (int d = 0; d < 4; ++d)
                tq[d] = sm[d * NPIX + p] * sm[(4 + d) * NPIX + p];
#pragma unroll
            for (int c = 0; c < 4; ++c) {
                float s = P.a1b[j][c];
#pragma unroll
                for (int d = 0; d < 4; ++d) s += P.a1w[j][c * 4 + d] * tq[d];
                a1[c] = s / (1.f + expf(-s)) * s == s ? s * (1.f / (1.f + expf(-s)))
                                                      : s;  // swish
                a1[c] = s * (1.f / (1.f + expf(-s)));
            }
#pragma unroll
            for (int c = 0; c < 4; ++c) {
                float s = P.a2b[j][c];
#pragma unroll
                for (int d = 0; d < 4; ++d) s += P.a2w[j][c * 4 + d] * a1[d];
                a2[c] = s;
            }
#pragma unroll
            for (int c = 0; c < 4; ++c)
                sx[(4 * j + c) * NPIX + p] =
                    tanhf(a2[c] * scal) * sm[(8 + c) * NPIX + p];
        }
        __syncthreads();
    }

    // ---- global branch: gq via WMMA (4 rows), 8x8 pooled K/V (9 keys) ----
    for (int tile = wave; tile < 36; tile += 8) {
        v8f c = {};
        c = wmma_gemm(P.gq_w, 16, 4, 16, sy + tile * 16, NPIX, 16, c);
        wmma_store_bm(sq + tile * 16, NPIX, 4, c, P.gq_b);
    }
    __syncthreads();
    // pool: sm[c*9+blk] = mean over 8x8 block of sy
    for (int idx = tid; idx < 16 * 9; idx += 256) {
        const int c = idx / 9, blk = idx % 9, bh = blk / 3, bw = blk % 3;
        float s = 0.f;
        for (int i = 0; i < 8; ++i)
            for (int jj = 0; jj < 8; ++jj)
                s += sy[c * NPIX + (bh * 8 + i) * WWID + bw * 8 + jj];
        sm[idx] = s * (1.f / 64.f);
    }
    __syncthreads();
    float* skv = sm + 256;  // 8x9: rows 0-3 = k(g,d), rows 4-7 = v(g,d)
    for (int idx = tid; idx < 8 * 9; idx += 256) {
        const int oc = idx / 9, blk = idx % 9;
        float s = P.gkv_b[oc];
        for (int kq = 0; kq < 16; ++kq) s += P.gkv_w[oc * 16 + kq] * sm[kq * 9 + blk];
        skv[idx] = s;
    }
    __syncthreads();
    // per-pixel 9-key attention (VALU)
    for (int p = tid; p < NPIX; p += 256) {
        for (int g = 0; g < 2; ++g) {
            const float q0 = sq[(g * 2 + 0) * NPIX + p];
            const float q1 = sq[(g * 2 + 1) * NPIX + p];
            float l[9], mx = -1e30f;
#pragma unroll
            for (int blk = 0; blk < 9; ++blk) {
                l[blk] = scal * (q0 * skv[(g * 2 + 0) * 9 + blk] +
                                 q1 * skv[(g * 2 + 1) * 9 + blk]);
                mx = fmaxf(mx, l[blk]);
            }
            float den = 0.f, r0 = 0.f, r1 = 0.f;
#pragma unroll
            for (int blk = 0; blk < 9; ++blk) {
                const float e = expf(l[blk] - mx);
                den += e;
                r0 += e * skv[(4 + g * 2 + 0) * 9 + blk];
                r1 += e * skv[(4 + g * 2 + 1) * 9 + blk];
            }
            const float inv = 1.f / den;
            sx[(12 + g * 2 + 0) * NPIX + p] = r0 * inv;
            sx[(12 + g * 2 + 1) * NPIX + p] = r1 * inv;
        }
    }
    __syncthreads();

    // ---- proj: out = proj_w(16x16) @ res(16x576) + b ----
    for (int tile = wave; tile < 36; tile += 8) {
        v8f c = {};
        c = wmma_gemm(P.proj_w, 16, 16, 16, sx + tile * 16, NPIX, 16, c);
        wmma_store_bm(out + (size_t)f * 16 * NPIX + tile * 16, NPIX, 16, c,
                      P.proj_b);
    }
}

// ============================================================================
// transformer: one sequence (T x 16) per wave; all linears + per-head QK^T and
// attn*V via WMMA; LN / softmax / pos-emb lane-per-row.
// layout in/out: elem (seq,t,d) at [((pb*T + t)*16 + d)*576 + pix], seq=pb*576+pix
// ============================================================================
struct TfP {
    const float *ln1g, *ln1b, *pew, *peb, *ln2g, *ln2b, *alng, *alnb;
    const float *qkvw, *ow, *ob, *flng, *flnb, *fw1, *fb1, *fw2, *fb2;
};

template <int T>
__device__ __forceinline__ void ln16_row(const float* in, float* out,
                                         const float* g, const float* b,
                                         int lane) {
    if (lane < T) {
        float v[16], m = 0.f;
#pragma unroll
        for (int d = 0; d < 16; ++d) { v[d] = in[lane * 16 + d]; m += v[d]; }
        m *= (1.f / 16.f);
        float var = 0.f;
#pragma unroll
        for (int d = 0; d < 16; ++d) { const float dd = v[d] - m; var += dd * dd; }
        const float inv = rsqrtf(var * (1.f / 16.f) + 1e-5f);
#pragma unroll
        for (int d = 0; d < 16; ++d)
            out[lane * 16 + d] = (v[d] - m) * inv * g[d] + b[d];
    }
}

template <int T, int WAVES>
__global__ __launch_bounds__(WAVES * 32) void tf_kernel(
    const float* __restrict__ in, float* __restrict__ outp, TfP P, int nseq) {
    constexpr int R = T / 16;
    constexpr int PER = T * 16 * 2 + T * 192 + T * T + T * 64;
    extern __shared__ float smem[];
    const int wave = threadIdx.x >> 5, lane = threadIdx.x & 31;
    float* sx   = smem + wave * PER;  // T x 16 residual stream
    float* sh   = sx + T * 16;        // T x 16 LN scratch
    float* sqkv = sh + T * 16;        // T x 192 (later T x 64 mlp hidden)
    float* sl   = sqkv + T * 192;     // T x T logits
    float* so   = sl + T * T;         // T x 64 attn output

    const int seq = blockIdx.x * WAVES + wave;
    const bool act = seq < nseq;
    const int pb = act ? seq / NPIX : 0;
    const int pix = act ? seq % NPIX : 0;
    const size_t fb = (size_t)pb * T;
    const int n16 = lane & 15, half = lane >> 4;

    if (lane < T) {
#pragma unroll
        for (int d = 0; d < 16; ++d)
            sx[lane * 16 + d] = act ? in[((fb + lane) * 16 + d) * NPIX + pix] : 0.f;
    }
    __syncthreads();
    // pe: LN1 -> linear -> LN2 -> +pos_emb
    ln16_row<T>(sx, sh, P.ln1g, P.ln1b, lane);
    __syncthreads();
    for (int r = 0; r < R; ++r) {
        v8f c = {};
        c = wmma_gemmT(sh + r * 256, 16, 16, 16, P.pew, 16, c);
        wmma_store_bn(sx + r * 256, 16, 16, c, P.peb);
    }
    __syncthreads();
    if (lane < T) {
        float v[16], m = 0.f;
#pragma unroll
        for (int d = 0; d < 16; ++d) { v[d] = sx[lane * 16 + d]; m += v[d]; }
        m *= (1.f / 16.f);
        float var = 0.f;
#pragma unroll
        for (int d = 0; d < 16; ++d) { const float dd = v[d] - m; var += dd * dd; }
        const float inv = rsqrtf(var * (1.f / 16.f) + 1e-5f);
#pragma unroll
        for (int d = 0; d < 16; ++d) {
            const float ln = (v[d] - m) * inv * P.ln2g[d] + P.ln2b[d];
            const int i2 = d >> 1;
            const float div = expf(-(float)(2 * i2) * 0.57564627324851142f);
            const float ang = (float)lane * div;
            sx[lane * 16 + d] = ln + ((d & 1) ? cosf(ang) : sinf(ang));
        }
    }
    __syncthreads();
    // attention block
    ln16_row<T>(sx, sh, P.alng, P.alnb, lane);
    __syncthreads();
    for (int ct = 0; ct < 12; ++ct)  // qkv: (Tx16) @ (16x192)
        for (int r = 0; r < R; ++r) {
            v8f c = {};
            c = wmma_gemmT(sh + r * 256, 16, 16, 16, P.qkvw + ct * 256, 16, c);
            wmma_store_bn(sqkv + r * 16 * 192 + ct * 16, 192, 16, c, nullptr);
        }
    __syncthreads();
    const float scal8 = 0.35355339059327373f;  // 8^-0.5
    for (int h8 = 0; h8 < 8; ++h8) {
        for (int ri = 0; ri < R; ++ri)  // logits = q_h @ k_h^T (K=8)
            for (int rj = 0; rj < R; ++rj) {
                v8f c = {};
                c = wmma_gemmT(sqkv + ri * 16 * 192 + h8 * 8, 192, 16, 8,
                               sqkv + rj * 16 * 192 + 64 + h8 * 8, 192, c);
                wmma_store_scale(sl + ri * 16 * T + rj * 16, T, c, scal8);
            }
        __syncthreads();
        if (lane < T) {  // softmax row
            float mx = -1e30f;
            for (int jj = 0; jj < T; ++jj) mx = fmaxf(mx, sl[lane * T + jj]);
            float s = 0.f;
            for (int jj = 0; jj < T; ++jj) {
                const float e = expf(sl[lane * T + jj] - mx);
                sl[lane * T + jj] = e;
                s += e;
            }
            const float inv = 1.f / s;
            for (int jj = 0; jj < T; ++jj) sl[lane * T + jj] *= inv;
        }
        __syncthreads();
        for (int ri = 0; ri < R; ++ri) {  // O_h = L @ v_h (N=8)
            v8f c = {};
            for (int rk = 0; rk < R; ++rk)
                c = wmma_gemm(sl + ri * 16 * T + rk * 16, T, 16, 16,
                              sqkv + rk * 16 * 192 + 128 + h8 * 8, 192, 8, c);
            wmma_store_bn(so + ri * 16 * 64 + h8 * 8, 64, 8, c, nullptr);
        }
        __syncthreads();
    }
    // o projection (K=64, 4 chained WMMA) + residual
    for (int r = 0; r < R; ++r) {
        v8f c = {};
        for (int kt = 0; kt < 4; ++kt)
            c = wmma_gemmT(so + r * 16 * 64 + kt * 16, 64, 16, 16, P.ow + kt * 16,
                           64, c);
#pragma unroll
        for (int rr = 0; rr < 8; ++rr) {
            const int m = r * 16 + rr + 8 * half;
            sx[m * 16 + n16] += c[rr] + P.ob[n16];
        }
    }
    __syncthreads();
    // MLP: f_ln -> fw1 -> gelu -> fw2 + residual -> global out
    ln16_row<T>(sx, sh, P.flng, P.flnb, lane);
    __syncthreads();
    for (int ct = 0; ct < 4; ++ct)
        for (int r = 0; r < R; ++r) {
            v8f c = {};
            c = wmma_gemmT(sh + r * 256, 16, 16, 16, P.fw1 + ct * 256, 16, c);
#pragma unroll
            for (int rr = 0; rr < 8; ++rr) {
                const int m = r * 16 + rr + 8 * half;
                const float v = c[rr] + P.fb1[ct * 16 + n16];
                sqkv[m * 64 + ct * 16 + n16] = gelu_exact(v);
            }
        }
    __syncthreads();
    for (int r = 0; r < R; ++r) {
        v8f c = {};
        for (int kt = 0; kt < 4; ++kt)
            c = wmma_gemmT(sqkv + r * 16 * 64 + kt * 16, 64, 16, 16,
                           P.fw2 + kt * 16, 64, c);
        if (act) {
#pragma unroll
            for (int rr = 0; rr < 8; ++rr) {
                const int tR = r * 16 + rr + 8 * half;
                outp[((fb + tR) * 16 + n16) * NPIX + pix] =
                    sx[tR * 16 + n16] + c[rr] + P.fb2[n16];
            }
        }
    }
}

// ============================================================================
// fusion: unpiece + concat(48) + LN + (48->1) + GELU
// ============================================================================
__global__ __launch_bounds__(256) void fusion_kernel(
    const float* __restrict__ xg, const float* __restrict__ pl,
    const float* __restrict__ pg, const float* __restrict__ g,
    const float* __restrict__ bln, const float* __restrict__ w,
    const float* __restrict__ bw, float* __restrict__ out, int total) {
    const int idx = blockIdx.x * 256 + threadIdx.x;
    if (idx >= total) return;
    const int bI = idx / (TFULL * NPIX);
    const int r = idx % (TFULL * NPIX);
    const int t = r / NPIX, p = r % NPIX;
    float v[48];
    const size_t fx = (size_t)(bI * TFULL + t);
#pragma unroll
    for (int c = 0; c < 16; ++c) v[c] = xg[(fx * 16 + c) * NPIX + p];
    size_t fp;
    if (t < WINL - 1) fp = (size_t)bI * WINL + t;                       // head
    else fp = ((size_t)(t - (WINL - 1)) * BB + bI) * WINL + (WINL - 1); // tail
#pragma unroll
    for (int c = 0; c < 16; ++c) {
        v[16 + c] = pl[(fp * 16 + c) * NPIX + p];
        v[32 + c] = pg[(fp * 16 + c) * NPIX + p];
    }
    float m = 0.f;
    for (int i = 0; i < 48; ++i) m += v[i];
    m *= (1.f / 48.f);
    float var = 0.f;
    for (int i = 0; i < 48; ++i) { const float d = v[i] - m; var += d * d; }
    const float inv = rsqrtf(var * (1.f / 48.f) + 1e-5f);
    float acc = bw[0];
    for (int i = 0; i < 48; ++i) acc += ((v[i] - m) * inv * g[i] + bln[i]) * w[i];
    out[idx] = gelu_exact(acc);
}

// ============================================================================
// host launcher
// ============================================================================
extern "C" void kernel_launch(void* const* d_in, const int* in_sizes, int n_in,
                              void* d_out, int out_size, void* d_ws,
                              size_t ws_size, hipStream_t stream) {
    (void)in_sizes; (void)n_in; (void)out_size; (void)ws_size;
    const float* const* in = (const float* const*)d_in;
    const float* x = in[0];

    auto mkAc = [&](int base) {
        AcP P;
        P.emb_w = in[base + 0];
        for (int j = 0; j < 3; ++j) {
            const int o = base + 1 + j * 8;
            P.qkv_w[j] = in[o + 0]; P.qkv_b[j] = in[o + 1];
            P.mix_w[j] = in[o + 2]; P.mix_b[j] = in[o + 3];
            P.a1w[j]   = in[o + 4]; P.a1b[j]   = in[o + 5];
            P.a2w[j]   = in[o + 6]; P.a2b[j]   = in[o + 7];
        }
        P.gq_w = in[base + 25];  P.gq_b = in[base + 26];
        P.gkv_w = in[base + 27]; P.gkv_b = in[base + 28];
        P.proj_w = in[base + 29]; P.proj_b = in[base + 30];
        return P;
    };
    auto mkTf = [&](int b) {
        TfP P;
        P.ln1g = in[b + 0];  P.ln1b = in[b + 1];
        P.pew  = in[b + 2];  P.peb  = in[b + 3];
        P.ln2g = in[b + 4];  P.ln2b = in[b + 5];
        P.alng = in[b + 6];  P.alnb = in[b + 7];
        P.qkvw = in[b + 8];
        P.ow   = in[b + 9];  P.ob   = in[b + 10];
        P.flng = in[b + 11]; P.flnb = in[b + 12];
        P.fw1  = in[b + 13]; P.fb1  = in[b + 14];
        P.fw2  = in[b + 15]; P.fb2  = in[b + 16];
        return P;
    };
    AcP localP = mkAc(1), globalP = mkAc(32);
    TfP ltt = mkTf(63), stt = mkTf(80);
    const float* fus_g = in[97];
    const float* fus_b = in[98];
    const float* fus_w = in[99];
    const float* fus_bias = in[100];

    // workspace layout (floats)
    const size_t F_XG = (size_t)BB * TFULL * 16 * NPIX;          // 589824
    const size_t F_P  = (size_t)NWINP * BB * WINL * 16 * NPIX;   // 5013504
    float* ws = (float*)d_ws;
    float* xg_conv = ws;
    float* xg_tf   = xg_conv + F_XG;
    float* pl_conv = xg_tf + F_XG;
    float* pg_conv = pl_conv + F_P;
    float* pl_tf   = pg_conv + F_P;
    float* pg_tf   = pl_tf + F_P;

    const size_t acLds = (size_t)(16 + 16 + 12 + 12) * NPIX * sizeof(float);
    const size_t tfLds16 = (size_t)8 * (16 * 288 + 16 * 16) * sizeof(float);
    const size_t tfLds32 = (size_t)4 * (32 * 288 + 32 * 32) * sizeof(float);

    // conv stages
    attnconv_kernel<0><<<BB * TFULL, 256, acLds, stream>>>(x, xg_conv, localP, 3, 3, 3);
    attnconv_kernel<1><<<NWINP * BB * WINL, 256, acLds, stream>>>(x, pl_conv, localP, 3, 3, 3);
    attnconv_kernel<1><<<NWINP * BB * WINL, 256, acLds, stream>>>(x, pg_conv, globalP, 7, 5, 3);

    // transformer stages
    const int nseq_x = BB * NPIX;               // 1152 sequences, T=32
    tf_kernel<32, 4><<<(nseq_x + 3) / 4, 128, tfLds32, stream>>>(xg_conv, xg_tf, ltt, nseq_x);
    const int nseq_p = NWINP * BB * NPIX;       // 19584 sequences, T=16
    tf_kernel<16, 8><<<(nseq_p + 7) / 8, 256, tfLds16, stream>>>(pl_conv, pl_tf, stt, nseq_p);
    tf_kernel<16, 8><<<(nseq_p + 7) / 8, 256, tfLds16, stream>>>(pg_conv, pg_tf, stt, nseq_p);

    // fusion
    const int total = BB * TFULL * NPIX;
    fusion_kernel<<<(total + 255) / 256, 256, 0, stream>>>(
        xg_tf, pl_tf, pg_tf, fus_g, fus_b, fus_w, fus_bias, (float*)d_out, total);
}
